// SimCLR_Loss_20117626815063
// MI455X (gfx1250) — matrix operand order repaired and checked by reading, not compiled, not run on hardware
//
#include <hip/hip_runtime.h>
#include <hip/hip_bf16.h>

typedef __attribute__((ext_vector_type(16))) __bf16 v16bf;
typedef __attribute__((ext_vector_type(8)))  __bf16 v8bf;
typedef __attribute__((ext_vector_type(8)))  float  v8f;

#define BATCH   4096
#define NROWS   8192
#define DDIM    128
#define TILE    16
#define NTILES  (NROWS / TILE)     /* 512 column tiles */
#define NSLICE  8                  /* column slices (waves per block) */
#define SLICET  (NTILES / NSLICE)  /* 64 tiles per slice */

// exp(2*x) == exp2(x * 2*log2(e)); v_exp_f32 is a raw exp2.
#define EXP2K   2.885390081777927f
#define TSCALE  2.0f               /* 1/temperature */

__device__ __forceinline__ unsigned short f2bf_rne(float f) {
    unsigned int u = __float_as_uint(f);
    u += 0x7fffu + ((u >> 16) & 1u);         // round-to-nearest-even
    return (unsigned short)(u >> 16);
}

// ---------------------------------------------------------------------------
// Kernel 1: z = [z_i ; z_j], row-normalize (eps=1e-8), emit bf16; zero d_out.
// ---------------------------------------------------------------------------
__global__ void simclr_normalize_kernel(const float* __restrict__ zi,
                                        const float* __restrict__ zj,
                                        unsigned short* __restrict__ znb,
                                        float* __restrict__ out)
{
    if (blockIdx.x == 0 && threadIdx.x == 0) out[0] = 0.0f;

    int wave = (int)((blockIdx.x * blockDim.x + threadIdx.x) >> 5);
    int lane = (int)(threadIdx.x & 31);
    if (wave >= NROWS) return;

    const float* src = (wave < BATCH) ? (zi + (size_t)wave * DDIM)
                                      : (zj + (size_t)(wave - BATCH) * DDIM);
    float4 v = ((const float4*)src)[lane];
    float ss = v.x * v.x + v.y * v.y + v.z * v.z + v.w * v.w;
    #pragma unroll
    for (int off = 16; off >= 1; off >>= 1) ss += __shfl_xor(ss, off, 32);

    float inv = 1.0f / fmaxf(sqrtf(ss), 1e-8f);

    unsigned int b0 = f2bf_rne(v.x * inv);
    unsigned int b1 = f2bf_rne(v.y * inv);
    unsigned int b2 = f2bf_rne(v.z * inv);
    unsigned int b3 = f2bf_rne(v.w * inv);
    uint2 o;
    o.x = b0 | (b1 << 16);
    o.y = b2 | (b3 << 16);
    ((uint2*)(znb + (size_t)wave * DDIM))[lane] = o;
}

// ---------------------------------------------------------------------------
// Helpers
// ---------------------------------------------------------------------------
__device__ __forceinline__ void load_b(v16bf b[4], const __bf16* p) {
    #pragma unroll
    for (int kc = 0; kc < 4; ++kc)
        b[kc] = *(const v16bf*)(p + kc * 32);
}

__device__ __forceinline__ void mm_pair(v8f& c0, v8f& c1,
                                        const v16bf a0[4], const v16bf a1[4],
                                        const v16bf b[4]) {
    #pragma unroll
    for (int kc = 0; kc < 4; ++kc) {
        c0 = __builtin_amdgcn_wmma_f32_16x16x32_bf16(false, a0[kc], false, b[kc],
                                                     (short)0, c0, false, false);
        c1 = __builtin_amdgcn_wmma_f32_16x16x32_bf16(false, a1[kc], false, b[kc],
                                                     (short)0, c1, false, false);
    }
}

__device__ __forceinline__ void accum(float s0[8], float s1[8],
                                      const v8f& c0, const v8f& c1) {
    #pragma unroll
    for (int v = 0; v < 8; ++v) {
        s0[v] += __builtin_amdgcn_exp2f(c0[v] * EXP2K);
        s1[v] += __builtin_amdgcn_exp2f(c1[v] * EXP2K);
    }
}

// Recompute one 16x16 S-tile (K=128) for a given column tile (cold path).
__device__ __forceinline__ v8f tile_mm(const v16bf a[4], const __bf16* znb,
                                       int ct, int l16, int half)
{
    const __bf16* bbase = znb + (size_t)(ct * TILE + l16) * DDIM + half * 16;
    v8f c = {};
    #pragma unroll
    for (int kc = 0; kc < 4; ++kc) {
        v16bf b = *(const v16bf*)(bbase + kc * 32);
        c = __builtin_amdgcn_wmma_f32_16x16x32_bf16(false, a[kc], false, b,
                                                    (short)0, c, false, false);
    }
    return c;
}

// Load the A fragment set (4 K-chunks of 32) for one 16-row tile.
__device__ __forceinline__ void load_a(v16bf a[4], const __bf16* znb,
                                       int rowTile, int l16, int half)
{
    const __bf16* abase = znb + (size_t)(rowTile * TILE + l16) * DDIM;
    #pragma unroll
    for (int kc = 0; kc < 4; ++kc) {
        v8bf lo = *(const v8bf*)(abase + kc * 32 + half * 16);
        v8bf hi = *(const v8bf*)(abase + kc * 32 + half * 16 + 8);
        a[kc] = __builtin_shufflevector(lo, hi,
                                        0, 1, 2, 3, 4, 5, 6, 7,
                                        8, 9, 10, 11, 12, 13, 14, 15);
    }
}

// ---------------------------------------------------------------------------
// Kernel 2: block = 8 waves = one 32-row group (two 16-row tiles per wave),
// each wave owns one of 8 column slices (64 tiles). Per tile: one 8-load
// clause fills b[4], WMMAs run behind short waits, 16 exps accumulate; the
// compiler interleaves two tiles per unrolled iteration. 2048 waves total
// give >=2 waves/SIMD so load-waits hide behind the co-resident wave.
// Diag/pair tiles peeled; LDS merges slices; one atomic per block.
// ---------------------------------------------------------------------------
__global__ void simclr_wmma_kernel(const unsigned short* __restrict__ znb_u16,
                                   float* __restrict__ out)
{
    __shared__ float lds_partial[NSLICE * 32];   // [slice][rowInGroup]
    __shared__ float lds_pos[32];                // positive logit per row

    const int group = (int)blockIdx.x;           // 0..255 : rows group*32..+32
    const int slice = (int)(threadIdx.x >> 5);   // 0..7   : column slice
    const int lane  = (int)(threadIdx.x & 31);
    const int half  = lane >> 4;
    const int l16   = lane & 15;

    const int rt0 = group * 2;                   // row tiles handled by wave
    const int rt1 = group * 2 + 1;

    const __bf16* znb = (const __bf16*)znb_u16;

    v16bf a0[4], a1[4];
    load_a(a0, znb, rt0, l16, half);
    load_a(a1, znb, rt1, l16, half);

    float s0[8], s1[8];
    #pragma unroll
    for (int v = 0; v < 8; ++v) { s0[v] = 0.0f; s1[v] = 0.0f; }

    const size_t bstride = (size_t)TILE * DDIM;
    const __bf16* bp = znb + (size_t)(slice * SLICET * TILE + l16) * DDIM
                           + half * 16;

    for (int i = 0; i < SLICET; ++i) {
        v16bf b[4];
        load_b(b, bp);                           // one 8-load clause
        v8f c0 = {}, c1 = {};
        mm_pair(c0, c1, a0, a1, b);              // 8 WMMAs
        accum(s0, s1, c0, c1);                   // 16 exps; next loads hoist here
        bp += bstride;
    }

    // ---- peel: subtract the true-diagonal exp (owning slice only) ---------
    #pragma unroll
    for (int t = 0; t < 2; ++t) {
        const int rt = t ? rt1 : rt0;
        if ((rt / SLICET) == slice) {            // diag tile ct == rt lives here
            v8f c = tile_mm(t ? a1 : a0, znb, rt, l16, half);
            #pragma unroll
            for (int v = 0; v < 8; ++v) {
                if (l16 == v + 8 * half) {       // tile-diagonal element
                    float e = __builtin_amdgcn_exp2f(c[v] * EXP2K);
                    if (t) s1[v] -= e; else s0[v] -= e;
                }
            }
        }
    }

    // ---- peel: extract positive-pair logit (owning slice only) ------------
    #pragma unroll
    for (int t = 0; t < 2; ++t) {
        const int rt = t ? rt1 : rt0;
        const int pt = (rt + (BATCH / TILE)) & (NTILES - 1);   // +-BATCH mod N
        if ((pt / SLICET) == slice) {
            v8f c = tile_mm(t ? a1 : a0, znb, pt, l16, half);
            #pragma unroll
            for (int v = 0; v < 8; ++v) {
                if (l16 == v + 8 * half)
                    lds_pos[t * 16 + v + 8 * half] = c[v] * TSCALE;
            }
        }
    }

    // ---- reduce each row's sum across its 16 lanes; deposit slice partial -
    #pragma unroll
    for (int t = 0; t < 2; ++t) {
        #pragma unroll
        for (int v = 0; v < 8; ++v) {
            float r = t ? s1[v] : s0[v];
            r += __shfl_xor(r, 1, 32);
            r += __shfl_xor(r, 2, 32);
            r += __shfl_xor(r, 4, 32);
            r += __shfl_xor(r, 8, 32);           // halves stay independent
            if (l16 == v + 8 * half)
                lds_partial[slice * 32 + t * 16 + v + 8 * half] = r;
        }
    }
    __syncthreads();

    // ---- wave 0: finish 32 rows, one atomic per block ---------------------
    if (threadIdx.x < 32) {
        int r = (int)threadIdx.x;
        float total = 0.0f;
        #pragma unroll
        for (int sl = 0; sl < NSLICE; ++sl)
            total += lds_partial[sl * 32 + r];
        float val = __logf(total) - lds_pos[r];  // lse_{c!=r} - pos_r
        #pragma unroll
        for (int off = 16; off >= 1; off >>= 1) val += __shfl_xor(val, off, 32);
        if (threadIdx.x == 0)
            atomicAdd(out, val * (1.0f / (float)NROWS));
    }
}

// ---------------------------------------------------------------------------
extern "C" void kernel_launch(void* const* d_in, const int* in_sizes, int n_in,
                              void* d_out, int out_size, void* d_ws, size_t ws_size,
                              hipStream_t stream) {
    (void)in_sizes; (void)n_in; (void)out_size; (void)ws_size;
    const float* zi = (const float*)d_in[0];
    const float* zj = (const float*)d_in[1];
    float* out = (float*)d_out;
    unsigned short* znb = (unsigned short*)d_ws;   // NROWS*DDIM bf16 = 2 MB

    simclr_normalize_kernel<<<NROWS / 8, 256, 0, stream>>>(zi, zj, znb, out);
    // 256 blocks x 8 waves: 32-row group per block, 8 column slices.
    simclr_wmma_kernel<<<256, NSLICE * 32, 0, stream>>>(znb, out);
}
